// RoIDelta_25048249270466
// MI455X (gfx1250) — matrix operand order repaired and checked by reading, not compiled, not run on hardware
//
#include <hip/hip_runtime.h>
#include <stdint.h>

// ---------------- problem constants (from the reference) ----------------
#define BB 64
#define NN 4096
#define MM 128
#define LL 21
#define TOTAL_POS 128
#define KPAD 136  // 128 + 8 halves row padding: 272B stride -> conflict-free, 16B aligned

typedef __attribute__((ext_vector_type(16))) _Float16     v16h;
typedef __attribute__((ext_vector_type(8)))  _Float16     v8h;
typedef __attribute__((ext_vector_type(8)))  float        v8f;
typedef __attribute__((ext_vector_type(4)))  float        v4f;
typedef __attribute__((ext_vector_type(4)))  unsigned int u32x4;
typedef __attribute__((ext_vector_type(8)))  int          i32x8;
typedef __attribute__((ext_vector_type(4)))  int          i32x4;

// =======================================================================
// Kernel 1: streaming zero-fill of the 110MB output (bandwidth bound:
// ~4.7us at 23.3 TB/s; this dominates the whole op).
// =======================================================================
__global__ void __launch_bounds__(256) zero_fill_kernel(v4f* __restrict__ out, int n4) {
  int i = blockIdx.x * 256 + threadIdx.x;
  if (i < n4) {
    v4f z = {0.f, 0.f, 0.f, 0.f};
    __builtin_nontemporal_store(z, &out[i]);
  }
}

// =======================================================================
// Kernel 2: fused IoU max/argmax. GT boxes (2KB/batch) staged in LDS via
// the CDNA5 Tensor Data Mover; IoU map [B,N,M] (128MB) never materialized.
// sel[b,n] = argmax_m IoU  if max IoU > 0.5, else -1.
// =======================================================================
__global__ void __launch_bounds__(256) iou_argmax_kernel(
    const float* __restrict__ roi, const float* __restrict__ gt,
    int* __restrict__ sel) {
  __shared__ float sgt[MM * 4];
  __shared__ float sarea[MM];

  const int b = blockIdx.x / (NN / 256);
  const int n = (blockIdx.x % (NN / 256)) * 256 + threadIdx.x;
  const float* gsrc = gt + (size_t)b * MM * 4;

#if __has_builtin(__builtin_amdgcn_tensor_load_to_lds)
  if (threadIdx.x < 32) {  // one wave issues the TDM descriptor (EXEC ignored by TDM)
    uint64_t gaddr = (uint64_t)(uintptr_t)gsrc;
    uint32_t laddr = (uint32_t)(uintptr_t)(&sgt[0]);  // low 32 bits = LDS byte addr
    // D# group 0: count=1, lds_addr, global_addr[56:0], type=2 ("image")
    u32x4 g0;
    g0[0] = 1u;
    g0[1] = laddr;
    g0[2] = (uint32_t)(gaddr & 0xFFFFFFFFu);
    g0[3] = (uint32_t)((gaddr >> 32) & 0x01FFFFFFu) | 0x80000000u;
    // D# group 1: data_size=4B, 1D tile of 512 contiguous f32 elements
    i32x8 g1 = {0, 0, 0, 0, 0, 0, 0, 0};
    g1[0] = (int)(2u << 16);                         // data_size = 4 bytes
    g1[1] = (int)(512u << 16);                       // tensor_dim0[15:0]  (bits 63:48)
    g1[2] = (int)(1u << 16);                         // tensor_dim0 hi=0, tensor_dim1=1
    g1[3] = (int)(512u << 16);                       // tile_dim0 = 512   (bits 127:112)
    g1[4] = 1;                                       // tile_dim1 = 1, tile_dim2 = 0
    g1[5] = 512;                                     // tensor_dim0_stride lo32
    g1[6] = (int)(512u << 16);                       // tensor_dim1_stride lo16
    i32x4 z4 = {0, 0, 0, 0};
#if __clang_major__ >= 23
    i32x8 z8 = {0, 0, 0, 0, 0, 0, 0, 0};
    __builtin_amdgcn_tensor_load_to_lds(g0, g1, z4, z4, z8, 0);
#else
    __builtin_amdgcn_tensor_load_to_lds(g0, g1, z4, z4, 0);
#endif
    __builtin_amdgcn_s_wait_tensorcnt(0);
  }
#else
  for (int i = threadIdx.x; i < MM * 4; i += 256) sgt[i] = gsrc[i];
#endif
  __syncthreads();

  if (threadIdx.x < MM) {
    float y1 = sgt[threadIdx.x * 4 + 0], x1 = sgt[threadIdx.x * 4 + 1];
    float y2 = sgt[threadIdx.x * 4 + 2], x2 = sgt[threadIdx.x * 4 + 3];
    sarea[threadIdx.x] = (y2 - y1) * (x2 - x1);
  }
  __syncthreads();

  const v4f rb = ((const v4f*)roi)[(size_t)b * NN + n];  // y1,x1,y2,x2
  const float by1 = rb.x, bx1 = rb.y, by2 = rb.z, bx2 = rb.w;
  const float barea = (by2 - by1) * (bx2 - bx1);

  // division-free running max: iou_m > best  <=>  inter*Ubest > Ibest*union
  float bestI = 0.f, bestU = 1.f;
  int bestM = 0;
#pragma unroll 4
  for (int m = 0; m < MM; ++m) {
    float gy1 = sgt[m * 4 + 0], gx1 = sgt[m * 4 + 1];
    float gy2 = sgt[m * 4 + 2], gx2 = sgt[m * 4 + 3];
    float xt = fmaxf(bx1, gx1), yt = fmaxf(by1, gy1);
    float xb = fminf(bx2, gx2), yb = fminf(by2, gy2);
    float iw = fmaxf(xb - xt, 0.f), ih = fmaxf(yb - yt, 0.f);
    float inter = iw * ih;
    float uni = barea + sarea[m] - inter;
    bool better = inter * bestU > bestI * uni;  // strict '>' keeps first max (jnp.argmax)
    bestI = better ? inter : bestI;
    bestU = better ? uni : bestU;
    bestM = better ? m : bestM;
  }
  // merged > 0.5  <=>  2*inter > union
  sel[(size_t)b * NN + n] = (2.f * bestI > bestU) ? bestM : -1;
}

// =======================================================================
// Kernel 3: per-batch-row selection (deterministic first-128 policy),
// WMMA one-hot gather of best GT box+label, delta computation, scatter.
// One 256-thread block (8 wave32) per batch row.
// =======================================================================
__global__ void __launch_bounds__(256) select_scatter_kernel(
    const float* __restrict__ roi, const float* __restrict__ gt,
    const int* __restrict__ lbl, const int* __restrict__ sel,
    float* __restrict__ out) {
  const int b = blockIdx.x;
  const int t = threadIdx.x;

  __shared__ float    sgt[MM * 4];
  __shared__ float    slbl[MM];
  __shared__ _Float16 sbh[16 * KPAD];       // B panel, hi f16 halves [N][K]
  __shared__ _Float16 sbl[16 * KPAD];       // B panel, lo f16 halves [N][K]
  __shared__ int      s_n[TOTAL_POS];
  __shared__ int      s_idx[TOTAL_POS];
  __shared__ float    s_g[TOTAL_POS * 16];  // gathered D tile rows (cols 0..4 used)
  __shared__ int      sc[256];
  __shared__ int      s_cnt;

  for (int i = t; i < MM * 4; i += 256) sgt[i] = gt[(size_t)b * MM * 4 + i];
  if (t < MM) slbl[t] = (float)lbl[b * MM + t];
  __syncthreads();

  // ---- precompute packed-f16 B panel: col N<4 = box coord (hi/lo split),
  //      N==4 = label (exact in f16), N>4 = 0. Row stride KPAD halves. ----
  for (int idx = t; idx < 16 * MM; idx += 256) {
    int N = idx >> 7;     // B column 0..15
    int K = idx & (MM - 1);
    float v = 0.f;
    if (N < 4)       v = sgt[K * 4 + N];
    else if (N == 4) v = slbl[K];
    _Float16 h = (_Float16)v;
    sbh[N * KPAD + K] = h;
    sbl[N * KPAD + K] = (_Float16)(v - (float)h);
  }

  // ---- block scan: keep the first TOTAL_POS positives per row ----
  const int base_n = t * 16;
  int flags[16];
  int c = 0;
  for (int i = 0; i < 16; ++i) {
    int s = sel[(size_t)b * NN + base_n + i];
    flags[i] = s;
    c += (s >= 0) ? 1 : 0;
  }
  sc[t] = c;
  __syncthreads();
  for (int off = 1; off < 256; off <<= 1) {
    int v = (t >= off) ? sc[t - off] : 0;
    __syncthreads();
    sc[t] += v;
    __syncthreads();
  }
  int rank = sc[t] - c;  // exclusive prefix
  if (t == 255) s_cnt = (sc[255] < TOTAL_POS) ? sc[255] : TOTAL_POS;
  for (int i = 0; i < 16; ++i) {
    if (flags[i] >= 0) {
      if (rank < TOTAL_POS) { s_n[rank] = base_n + i; s_idx[rank] = flags[i]; }
      ++rank;
    }
  }
  __syncthreads();
  const int cnt = s_cnt;
  if (t >= cnt && t < TOTAL_POS) { s_n[t] = -1; s_idx[t] = 0; }  // pad tiles
  __syncthreads();

  // ---- WMMA one-hot gather: D[16x16] = OneHot[16x128] @ GT[128x16] ----
  // A one-hot is exact in f16; box coords split hi/lo f16 accumulated into
  // the same f32 C (~22-bit accurate); labels exact in f16.
  const int w = t >> 5, lane = t & 31;
  const int mrow = lane & 15;   // A tile row / B,D column this lane holds
  const int hi16 = lane >> 4;   // upper half-wave flag
  const int r0 = w * 16;        // 8 waves -> 8 tiles of 16 selected ROIs
  const int myidx = s_idx[r0 + mrow];
  const _Float16* bhrow = &sbh[mrow * KPAD];
  const _Float16* blrow = &sbl[mrow * KPAD];

  v8f acc = {0.f, 0.f, 0.f, 0.f, 0.f, 0.f, 0.f, 0.f};
#pragma unroll
  for (int ch = 0; ch < 4; ++ch) {  // K = 128 split into 4 chunks of 32
    // A 16x32 f16 lane layout: element i <-> local K = i + 16*hi16 (matches B)
    v16h a;
#pragma unroll
    for (int i = 0; i < 16; ++i) {
      int K = ch * 32 + 16 * hi16 + i;
      a[i] = (K == myidx) ? (_Float16)1.0f : (_Float16)0.0f;
    }
    // B 32x16 f16: element i <-> local K = i + 16*hi16, column N = mrow.
    // Contiguous 16-half slices -> two ds_load_b128 each, branch-free.
    const int kb = ch * 32 + 16 * hi16;
    v8h h0 = *(const v8h*)(bhrow + kb);
    v8h h1 = *(const v8h*)(bhrow + kb + 8);
    v8h l0 = *(const v8h*)(blrow + kb);
    v8h l1 = *(const v8h*)(blrow + kb + 8);
    v16h bh = __builtin_shufflevector(h0, h1, 0, 1, 2, 3, 4, 5, 6, 7,
                                      8, 9, 10, 11, 12, 13, 14, 15);
    v16h bl = __builtin_shufflevector(l0, l1, 0, 1, 2, 3, 4, 5, 6, 7,
                                      8, 9, 10, 11, 12, 13, 14, 15);
    acc = __builtin_amdgcn_wmma_f32_16x16x32_f16(false, a, false, bh,
                                                 (short)0, acc, false, false);
    acc = __builtin_amdgcn_wmma_f32_16x16x32_f16(false, a, false, bl,
                                                 (short)0, acc, false, false);
  }
  // D layout: vgpr p, lane L -> D[M = p + 8*(L>=16)][N = L%16].
  // Unconditional write-back: every lane owns column 'mrow' of its rows.
#pragma unroll
  for (int p = 0; p < 8; ++p) s_g[(r0 + p + 8 * hi16) * 16 + mrow] = acc[p];
  __syncthreads();

  // ---- deltas / variances + scatter of the <=128 nonzero rows ----
  if (t < cnt) {
    int n = s_n[t];
    const v4f rbx = ((const v4f*)roi)[(size_t)b * NN + n];  // y1,x1,y2,x2
    float gy1 = s_g[t * 16 + 0], gx1 = s_g[t * 16 + 1];
    float gy2 = s_g[t * 16 + 2], gx2 = s_g[t * 16 + 3];
    int label = (int)(s_g[t * 16 + 4] + 0.5f);

    float bh_ = rbx.z - rbx.x, bw_ = rbx.w - rbx.y;
    float bcy = rbx.x + 0.5f * bh_, bcx = rbx.y + 0.5f * bw_;
    float gh_ = gy2 - gy1, gw_ = gx2 - gx1;
    float gcy = gy1 + 0.5f * gh_, gcx = gx1 + 0.5f * gw_;
    bw_ = (bw_ == 0.f) ? 1e-3f : bw_;
    bh_ = (bh_ == 0.f) ? 1e-3f : bh_;
    float dx = (gcx - bcx) / bw_;
    float dy = (gcy - bcy) / bh_;
    float dw = (gw_ == 0.f) ? 0.f : logf(gw_ / bw_);
    float dh = (gh_ == 0.f) ? 0.f : logf(gh_ / bh_);
    dy /= 0.1f; dx /= 0.1f; dh /= 0.2f; dw /= 0.2f;   // VARIANCES

    size_t row = (size_t)b * NN + n;
    float* dst = out + (row * LL + (size_t)label) * 4;
    dst[0] = dy; dst[1] = dx; dst[2] = dh; dst[3] = dw;
    out[(size_t)BB * NN * LL * 4 + row * LL + (size_t)label] = 1.0f;
  }
}

// =======================================================================
extern "C" void kernel_launch(void* const* d_in, const int* in_sizes, int n_in,
                              void* d_out, int out_size, void* d_ws, size_t ws_size,
                              hipStream_t stream) {
  const float* roi = (const float*)d_in[0];  // [B,N,4]
  const float* gt  = (const float*)d_in[1];  // [B,M,4]
  const int*   lbl = (const int*)d_in[2];    // [B,M]
  float* out = (float*)d_out;                // [B,N,21,4] ++ [B,N,21]
  int* sel = (int*)d_ws;                     // [B,N] scratch (1 MB)

  int n4 = out_size >> 2;  // out_size = 27,525,120, divisible by 4
  zero_fill_kernel<<<(n4 + 255) / 256, 256, 0, stream>>>((v4f*)out, n4);
  iou_argmax_kernel<<<BB * (NN / 256), 256, 0, stream>>>(roi, gt, sel);
  select_scatter_kernel<<<BB, 256, 0, stream>>>(roi, gt, lbl, sel, out);
}
